// RGAT_22333829939345
// MI455X (gfx1250) — compile-verified, matrix operand order checked by reading, not run on hardware
//
#include <hip/hip_runtime.h>
#include <hip/hip_bf16.h>

#define NNODES 50000
#define NEDGES 400000
#define RREL   3
#define FDIM   256   // feature dim at every GEMM stage (IN = H*HID = H*OUT = 256)
#define NHEADS 4
#define HDIM   64
#define NEG_SLOPE 0.2f

#define GEMM_BLK_X ((NNODES + 63) / 64)     // 782 blocks of 4 row-tiles
#define NROWS_PAD  (GEMM_BLK_X * 64)        // 50048 padded rows for A

typedef __attribute__((ext_vector_type(16))) __bf16 v16bf;
typedef __attribute__((ext_vector_type(8)))  __bf16 v8bf;
typedef __attribute__((ext_vector_type(8)))  float  v8f;

// ---------------------------------------------------------------------------
// helpers
// ---------------------------------------------------------------------------
__device__ __forceinline__ void atomicMaxF(float* addr, float v) {
    // monotonic float<->int trick; valid for mixed-sign values, init = -inf
    if (v >= 0.0f) atomicMax(reinterpret_cast<int*>(addr), __float_as_int(v));
    else           atomicMin(reinterpret_cast<unsigned int*>(addr), __float_as_uint(v));
}

__device__ __forceinline__ v16bf load_mat_frag(const __bf16* p) {
    // ISA 16-bit A/B fragment: elements 0..7 = K[sub*8 .. +7], 8..15 = K[sub*8+16 .. +23]
    v8bf lo = *reinterpret_cast<const v8bf*>(p);
    v8bf hi = *reinterpret_cast<const v8bf*>(p + 16);
    return __builtin_shufflevector(lo, hi, 0,1,2,3,4,5,6,7,8,9,10,11,12,13,14,15);
}

// ---------------------------------------------------------------------------
// conversion / prep kernels
// ---------------------------------------------------------------------------
__global__ void cvt_bf16(const float* __restrict__ src, __bf16* __restrict__ dstp,
                         float scale, int total) {
    int t = blockIdx.x * blockDim.x + threadIdx.x;
    if (t >= total) return;
    dstp[t] = (__bf16)(src[t] * scale);
}

__global__ void transpose_w(const float* __restrict__ W, __bf16* __restrict__ Wt) {
    // Wt[n][k] = W[k][n], 256x256
    int t = blockIdx.x * blockDim.x + threadIdx.x;   // t = n*256 + k
    int n = t >> 8, k = t & 255;
    Wt[t] = (__bf16)W[k * FDIM + n];
}

// ---------------------------------------------------------------------------
// WMMA GEMM: feat(N x 256) = h_bf16(Npad x 256) @ W(256 x 256)
// grid = (782, 4 heads) x 128 threads (4 waves).
// Block stages its head's 64x256 bf16 W panel (32 KB) in LDS; wave w owns
// row-tile m0 = (blockIdx.x*4 + w)*16 and all 64 cols of the head.
// ---------------------------------------------------------------------------
__global__ __launch_bounds__(128) void gemm_wmma(const __bf16* __restrict__ A,
                                                 const __bf16* __restrict__ Bt,  // Bt[n][k]
                                                 float* __restrict__ C) {
    __shared__ __bf16 lB[64 * FDIM];         // 32 KB B panel for this head
    const int tid  = threadIdx.x;
    const int wave = tid >> 5;
    const int lane = tid & 31;
    const int sub  = lane >> 4;              // K-octet selector
    const int l16  = lane & 15;              // M (A) / N (B) within tile
    const int head = blockIdx.y;
    const int m0   = (blockIdx.x * 4 + wave) * 16;

    // cooperative copy: 64*256 bf16 = 32 KB = 2048 x float4
    {
        const float4* s = reinterpret_cast<const float4*>(Bt + (size_t)head * 64 * FDIM);
        float4* d = reinterpret_cast<float4*>(lB);
#pragma unroll
        for (int i = 0; i < 16; ++i) d[tid + i * 128] = s[tid + i * 128];
    }
    __syncthreads();

    v8f acc0 = {}, acc1 = {}, acc2 = {}, acc3 = {};
    const __bf16* arow = A + (size_t)(m0 + l16) * FDIM + sub * 8;
    const __bf16* lb0  = lB + ( 0 + l16) * FDIM + sub * 8;
    const __bf16* lb1  = lB + (16 + l16) * FDIM + sub * 8;
    const __bf16* lb2  = lB + (32 + l16) * FDIM + sub * 8;
    const __bf16* lb3  = lB + (48 + l16) * FDIM + sub * 8;

#pragma unroll
    for (int k0 = 0; k0 < FDIM; k0 += 32) {
        v16bf a   = load_mat_frag(arow + k0);   // global_load_b128 x2
        v16bf vb0 = load_mat_frag(lb0 + k0);    // ds_load_b128 x2
        v16bf vb1 = load_mat_frag(lb1 + k0);
        v16bf vb2 = load_mat_frag(lb2 + k0);
        v16bf vb3 = load_mat_frag(lb3 + k0);
        acc0 = __builtin_amdgcn_wmma_f32_16x16x32_bf16(false, a, false, vb0, (short)0, acc0, false, false);
        acc1 = __builtin_amdgcn_wmma_f32_16x16x32_bf16(false, a, false, vb1, (short)0, acc1, false, false);
        acc2 = __builtin_amdgcn_wmma_f32_16x16x32_bf16(false, a, false, vb2, (short)0, acc2, false, false);
        acc3 = __builtin_amdgcn_wmma_f32_16x16x32_bf16(false, a, false, vb3, (short)0, acc3, false, false);
    }

    if (m0 < NNODES) {                       // wave-uniform guard (48-row overhang)
        // C/D layout: lane = N (0..15 twice), VGPR r -> M = r + 8*sub
        float* crow = C + (size_t)m0 * FDIM + head * 64 + l16;
#pragma unroll
        for (int r = 0; r < 8; ++r) {
            int m = r + sub * 8;
            crow[(size_t)m * FDIM +  0] = acc0[r];
            crow[(size_t)m * FDIM + 16] = acc1[r];
            crow[(size_t)m * FDIM + 32] = acc2[r];
            crow[(size_t)m * FDIM + 48] = acc3[r];
        }
    }
}

// ---------------------------------------------------------------------------
// attention logits: el[n,h] = <feat[n,h,:], al[h,:]>, er likewise
// one thread per (n,h); feat offset = t*64 since t = n*4+h
// ---------------------------------------------------------------------------
__global__ void attn_logits(const float* __restrict__ feat,
                            const float* __restrict__ al, const float* __restrict__ ar,
                            float* __restrict__ el, float* __restrict__ er, int total) {
    int t = blockIdx.x * blockDim.x + threadIdx.x;
    if (t >= total) return;
    int h = t & 3;
    const float4* f  = reinterpret_cast<const float4*>(feat + (size_t)t * HDIM);
    const float4* av = reinterpret_cast<const float4*>(al + h * HDIM);
    const float4* rv = reinterpret_cast<const float4*>(ar + h * HDIM);
    float sl = 0.f, sr = 0.f;
#pragma unroll
    for (int i = 0; i < HDIM / 4; ++i) {
        float4 fv = f[i], a = av[i], r = rv[i];
        sl += fv.x*a.x + fv.y*a.y + fv.z*a.z + fv.w*a.w;
        sr += fv.x*r.x + fv.y*r.y + fv.z*r.z + fv.w*r.w;
    }
    el[t] = sl; er[t] = sr;
}

__global__ void fill_softmax_state(float* __restrict__ emax, float* __restrict__ denom, int total) {
    int t = blockIdx.x * blockDim.x + threadIdx.x;
    if (t >= total) return;
    emax[t] = __int_as_float(0xff800000);  // -inf
    denom[t] = 0.0f;
}

// ---------------------------------------------------------------------------
// edge pass 1: e = leaky_relu(el[src]+er[dst]); segment max into emax[dst]
// ---------------------------------------------------------------------------
__global__ void edge_logits_max(const int* __restrict__ src, const int* __restrict__ dst,
                                const float* __restrict__ el, const float* __restrict__ er,
                                float* __restrict__ ebuf, float* __restrict__ emax, int total) {
    int t = blockIdx.x * blockDim.x + threadIdx.x;
    if (t >= total) return;
    int i = t >> 2, h = t & 3;
    int s = src[i], d = dst[i];
    float v = el[s * NHEADS + h] + er[d * NHEADS + h];
    v = v > 0.0f ? v : NEG_SLOPE * v;
    ebuf[t] = v;
    atomicMaxF(emax + d * NHEADS + h, v);
}

// edge pass 2: ex = exp(e - emax[dst]); segment sum into denom[dst]
__global__ void edge_exp_sum(const int* __restrict__ dst, float* __restrict__ ebuf,
                             const float* __restrict__ emax, float* __restrict__ denom, int total) {
    int t = blockIdx.x * blockDim.x + threadIdx.x;
    if (t >= total) return;
    int i = t >> 2, h = t & 3;
    int d = dst[i];
    float xv = __expf(ebuf[t] - emax[d * NHEADS + h]);
    ebuf[t] = xv;
    unsafeAtomicAdd(denom + d * NHEADS + h, xv);
}

// edge pass 3: one wave per edge; lane owns 8 of the 256 feature values.
// alpha = ex/denom per head (head = lane>>3); scatter-add msg into agg[dst].
__global__ __launch_bounds__(256) void edge_scatter(const int* __restrict__ src,
                                                    const int* __restrict__ dst,
                                                    const float* __restrict__ ebuf,
                                                    const float* __restrict__ denom,
                                                    const float* __restrict__ feat,
                                                    float* __restrict__ agg, int nedges) {
    int gid  = (blockIdx.x * blockDim.x + threadIdx.x) >> 5;
    int lane = threadIdx.x & 31;
    if (gid >= nedges) return;
    int s = src[gid], d = dst[gid];
    int h = lane >> 3;
    float alpha = ebuf[(size_t)gid * NHEADS + h] /
                  fmaxf(denom[d * NHEADS + h], 1e-9f);
    const float4* f = reinterpret_cast<const float4*>(feat + (size_t)s * FDIM + lane * 8);
    float4 f0 = f[0], f1 = f[1];
    float* ag = agg + (size_t)d * FDIM + lane * 8;
    unsafeAtomicAdd(ag + 0, f0.x * alpha);
    unsafeAtomicAdd(ag + 1, f0.y * alpha);
    unsafeAtomicAdd(ag + 2, f0.z * alpha);
    unsafeAtomicAdd(ag + 3, f0.w * alpha);
    unsafeAtomicAdd(ag + 4, f1.x * alpha);
    unsafeAtomicAdd(ag + 5, f1.y * alpha);
    unsafeAtomicAdd(ag + 6, f1.z * alpha);
    unsafeAtomicAdd(ag + 7, f1.w * alpha);
}

// accum += elu(agg + b); re-zero agg for the next relation (keeps one pass)
__global__ void relation_finalize(float* __restrict__ agg, float* __restrict__ accum,
                                  const float* __restrict__ b, int total) {
    int t = blockIdx.x * blockDim.x + threadIdx.x;
    if (t >= total) return;
    float v = agg[t] + b[t & (FDIM - 1)];
    accum[t] += (v > 0.0f) ? v : (__expf(v) - 1.0f);
    agg[t] = 0.0f;
}

// final: out[n,d] = (1/R) * mean_h accum[n,h,d] = sum_h accum / 12
__global__ void head_mean(const float* __restrict__ accum, float* __restrict__ out, int total) {
    int t = blockIdx.x * blockDim.x + threadIdx.x;
    if (t >= total) return;
    int n = t >> 6, d = t & (HDIM - 1);
    const float* a = accum + (size_t)n * FDIM + d;
    out[t] = (a[0] + a[64] + a[128] + a[192]) * (1.0f / 12.0f);
}

// ---------------------------------------------------------------------------
extern "C" void kernel_launch(void* const* d_in, const int* in_sizes, int n_in,
                              void* d_out, int out_size, void* d_ws, size_t ws_size,
                              hipStream_t stream) {
    (void)in_sizes; (void)n_in; (void)out_size; (void)ws_size;

    const float* x   = (const float*)d_in[0];
    const int*   src = (const int*)d_in[1];   // (R, E)
    const int*   dst = (const int*)d_in[2];   // (R, E)
    const float* Wl[3]  = {(const float*)d_in[3], (const float*)d_in[7],  (const float*)d_in[11]};
    const float* alL[3] = {(const float*)d_in[4], (const float*)d_in[8],  (const float*)d_in[12]};
    const float* arL[3] = {(const float*)d_in[5], (const float*)d_in[9],  (const float*)d_in[13]};
    const float* bL[3]  = {(const float*)d_in[6], (const float*)d_in[10], (const float*)d_in[14]};

    char* p = (char*)d_ws;
    auto carve = [&](size_t bytes) { char* r = p; p += (bytes + 255) & ~(size_t)255; return r; };
    __bf16* h_bf  = (__bf16*)carve((size_t)NROWS_PAD * FDIM * 2);     // 25.6 MB (padded rows)
    __bf16* Wt    = (__bf16*)carve((size_t)FDIM * FDIM * 2);          // 128 KB
    float*  feat  = (float*) carve((size_t)NNODES * FDIM * 4);        // 51.2 MB
    float*  el    = (float*) carve((size_t)NNODES * NHEADS * 4);
    float*  er    = (float*) carve((size_t)NNODES * NHEADS * 4);
    float*  emax  = (float*) carve((size_t)NNODES * NHEADS * 4);
    float*  denom = (float*) carve((size_t)NNODES * NHEADS * 4);
    float*  ebuf  = (float*) carve((size_t)NEDGES * NHEADS * 4);      // 6.4 MB
    float*  agg   = (float*) carve((size_t)NNODES * FDIM * 4);        // 51.2 MB
    float*  accum = (float*) carve((size_t)NNODES * FDIM * 4);        // 51.2 MB

    hipMemsetAsync(agg, 0, (size_t)NNODES * FDIM * 4, stream);

    const int TOT_NF = NNODES * FDIM;        // 12.8M
    const int TOT_NH = NNODES * NHEADS;      // 200k
    const int TOT_EH = NEDGES * NHEADS;      // 1.6M

    for (int li = 0; li < 3; ++li) {
        const float* hsrc = (li == 0) ? x : accum;
        float scale = (li == 0) ? 1.0f : (1.0f / (float)RREL);  // fold cross-relation mean
        cvt_bf16<<<TOT_NF / 256, 256, 0, stream>>>(hsrc, h_bf, scale, TOT_NF);
        hipMemsetAsync(accum, 0, (size_t)TOT_NF * 4, stream);

        for (int r = 0; r < RREL; ++r) {
            const int* sr = src + (size_t)r * NEDGES;
            const int* dr = dst + (size_t)r * NEDGES;
            transpose_w<<<FDIM * FDIM / 256, 256, 0, stream>>>(Wl[li] + (size_t)r * FDIM * FDIM, Wt);
            gemm_wmma<<<dim3(GEMM_BLK_X, NHEADS), 128, 0, stream>>>(h_bf, Wt, feat);
            attn_logits<<<(TOT_NH + 255) / 256, 256, 0, stream>>>(
                feat, alL[li] + r * FDIM, arL[li] + r * FDIM, el, er, TOT_NH);
            fill_softmax_state<<<(TOT_NH + 255) / 256, 256, 0, stream>>>(emax, denom, TOT_NH);
            edge_logits_max<<<TOT_EH / 256, 256, 0, stream>>>(sr, dr, el, er, ebuf, emax, TOT_EH);
            edge_exp_sum<<<TOT_EH / 256, 256, 0, stream>>>(dr, ebuf, emax, denom, TOT_EH);
            edge_scatter<<<NEDGES / 8, 256, 0, stream>>>(sr, dr, ebuf, denom, feat, agg, NEDGES);
            relation_finalize<<<TOT_NF / 256, 256, 0, stream>>>(agg, accum, bL[li] + r * FDIM, TOT_NF);
        }
    }
    head_mean<<<NNODES * HDIM / 256, 256, 0, stream>>>(accum, (float*)d_out, NNODES * HDIM);
}
